// KNeighborsVC_9895604650416
// MI455X (gfx1250) — compile-verified, matrix operand order checked by reading, not run on hardware
//
#include <hip/hip_runtime.h>
#include <hip/hip_bf16.h>

typedef __attribute__((ext_vector_type(16))) __bf16 v16bf;
typedef __attribute__((ext_vector_type(8)))  float  v8f;

#define NQ      2048
#define NM      100000
#define DIM     1024
#define QT      128      // query rows per block
#define NSTEP   256      // score columns per outer step (4 waves x 64)
#define CHUNK   2048     // matching columns per block
#define NCHUNK  49       // ceil(NM / CHUNK)
#define KT      64       // K step staged to LDS (2 WMMA K-blocks of 32)
#define TOPK    4
#define NEG_INF (-3.402823466e38f)

__device__ __forceinline__ unsigned short f2bf(float f) {
  unsigned int u = __float_as_uint(f);
  u += 0x7FFFu + ((u >> 16) & 1u);   // round-to-nearest-even
  return (unsigned short)(u >> 16);
}

__device__ __forceinline__ void top4_insert(float (&bv)[TOPK], int (&bi)[TOPK],
                                            float v, int i) {
  if (v <= bv[3]) return;
  if (v > bv[1]) {
    bv[3] = bv[2]; bi[3] = bi[2];
    bv[2] = bv[1]; bi[2] = bi[1];
    if (v > bv[0]) { bv[1] = bv[0]; bi[1] = bi[0]; bv[0] = v; bi[0] = i; }
    else           { bv[1] = v;     bi[1] = i; }
  } else {
    if (v > bv[2]) { bv[3] = bv[2]; bi[3] = bi[2]; bv[2] = v; bi[2] = i; }
    else           { bv[3] = v;     bi[3] = i; }
  }
}

// ---------------- one-time convert: Q (f32) -> Qb (bf16) -------------------
__global__ __launch_bounds__(256) void conv_q_kernel(const float* __restrict__ Q,
                                                     unsigned short* __restrict__ Qb) {
  const int wid  = threadIdx.x >> 5;
  const int lane = threadIdx.x & 31;
  const int row  = blockIdx.x * 8 + wid;           // 256 blocks * 8 = 2048
  const float4* p = reinterpret_cast<const float4*>(Q + (size_t)row * DIM);
#pragma unroll
  for (int i = 0; i < 8; ++i) {
    float4 v = p[lane + i * 32];
    union { unsigned short s4[4]; uint2 u; } t;
    t.s4[0] = f2bf(v.x); t.s4[1] = f2bf(v.y); t.s4[2] = f2bf(v.z); t.s4[3] = f2bf(v.w);
    *reinterpret_cast<uint2*>(Qb + (size_t)row * DIM + (lane + i * 32) * 4) = t.u;
  }
}

// ------ one-time: M (f32) -> row-normalized Mb (bf16), norms fused ---------
__global__ __launch_bounds__(256) void conv_m_kernel(const float* __restrict__ M,
                                                     unsigned short* __restrict__ Mb) {
  const int wid  = threadIdx.x >> 5;
  const int lane = threadIdx.x & 31;
  const int row  = blockIdx.x * 8 + wid;           // 12500 blocks * 8 = 100000
  if (row >= NM) return;
  const float4* p = reinterpret_cast<const float4*>(M + (size_t)row * DIM);
  float4 v[8];
  float s = 0.f;
#pragma unroll
  for (int i = 0; i < 8; ++i) {
    v[i] = p[lane + i * 32];
    s += v[i].x * v[i].x + v[i].y * v[i].y + v[i].z * v[i].z + v[i].w * v[i].w;
  }
#pragma unroll
  for (int off = 16; off; off >>= 1) s += __shfl_xor(s, off, 32);
  const float r = rsqrtf(s);
#pragma unroll
  for (int i = 0; i < 8; ++i) {
    union { unsigned short s4[4]; uint2 u; } t;
    t.s4[0] = f2bf(v[i].x * r); t.s4[1] = f2bf(v[i].y * r);
    t.s4[2] = f2bf(v[i].z * r); t.s4[3] = f2bf(v[i].w * r);
    *reinterpret_cast<uint2*>(Mb + (size_t)row * DIM + (lane + i * 32) * 4) = t.u;
  }
}

// ------------------------------------------ similarity GEMM + chunk top-4 --
__global__ __launch_bounds__(256) void simtopk_kernel(
    const unsigned short* __restrict__ Qb, const unsigned short* __restrict__ Mb,
    float* __restrict__ pval, int* __restrict__ pidx) {
  __shared__ __align__(16) unsigned short Atile[QT][KT];     // 16 KB
  __shared__ __align__(16) unsigned short Btile[NSTEP][KT];  // 32 KB
  __shared__ __align__(16) float Stile[QT][64];              // 32 KB

  const int tid   = threadIdx.x;
  const int lane  = tid & 31;
  const int wid   = tid >> 5;
  const int m_w   = wid & 1;          // 2 row-slices of 64
  const int n_w   = wid >> 1;         // 4 col-slices of 64
  const int qrow0 = blockIdx.x * QT;
  const int col0  = blockIdx.y * CHUNK;

  const int selRow  = tid >> 1;       // 0..127
  const int selHalf = tid & 1;        // 0/1 -> 32 columns each pass

  const int mrow  = lane & 15;
  const int khalf = (lane >> 4) * 8;

  float bv[TOPK] = {NEG_INF, NEG_INF, NEG_INF, NEG_INF};
  int   bi[TOPK] = {0, 0, 0, 0};

  // staging thread mapping
  const int ar = tid >> 1, ah = tid & 1;              // A: 2 threads/row, 64 B each
  const int br = tid;                                 // B: 1 thread/row, 128 B

  for (int ns = 0; ns < CHUNK / NSTEP; ++ns) {
    const int nbase = col0 + ns * NSTEP;
    const int gcolB = nbase + br;
    const unsigned short* aSrcRow = Qb + (size_t)(qrow0 + ar) * DIM + ah * 32;
    const unsigned short* bSrcRow = Mb + (size_t)gcolB * DIM;

    v8f acc[4][4] = {};
    uint4 pa[4], pb[8];

    // ---- prologue: prefetch K-step 0 ------------------------------------
    {
      const uint4* srcA = reinterpret_cast<const uint4*>(aSrcRow);
#pragma unroll
      for (int i = 0; i < 4; ++i) pa[i] = srcA[i];
#pragma unroll
      for (int i = 0; i < 8; ++i) pb[i] = make_uint4(0u, 0u, 0u, 0u);
      if (gcolB < NM) {
        const uint4* srcB = reinterpret_cast<const uint4*>(bSrcRow);
#pragma unroll
        for (int i = 0; i < 8; ++i) pb[i] = srcB[i];
      }
    }

    for (int kb = 0; kb < DIM; kb += KT) {
      __syncthreads();  // previous fragment / Stile reads done
      // ---- commit prefetched K-step to LDS -------------------------------
      {
        uint4* dstA = reinterpret_cast<uint4*>(&Atile[ar][ah * 32]);
#pragma unroll
        for (int i = 0; i < 4; ++i) dstA[i] = pa[i];
        uint4* dstB = reinterpret_cast<uint4*>(&Btile[br][0]);
#pragma unroll
        for (int i = 0; i < 8; ++i) dstB[i] = pb[i];
      }
      __syncthreads();

      // ---- prefetch next K-step (overlaps WMMA below) --------------------
      if (kb + KT < DIM) {
        const uint4* srcA = reinterpret_cast<const uint4*>(aSrcRow + kb + KT);
#pragma unroll
        for (int i = 0; i < 4; ++i) pa[i] = srcA[i];
        if (gcolB < NM) {
          const uint4* srcB = reinterpret_cast<const uint4*>(bSrcRow + kb + KT);
#pragma unroll
          for (int i = 0; i < 8; ++i) pb[i] = srcB[i];
        }
      }

      // ---- 2 WMMA K-blocks of 32 from this LDS stage ---------------------
#pragma unroll
      for (int kk = 0; kk < 2; ++kk) {
        const int koff = kk * 32 + khalf;
        v16bf b[4];
#pragma unroll
        for (int ni = 0; ni < 4; ++ni) {
          union { v16bf v; uint4 u[2]; } un;
          const unsigned short* p = &Btile[n_w * 64 + ni * 16 + mrow][koff];
          un.u[0] = *reinterpret_cast<const uint4*>(p);
          un.u[1] = *reinterpret_cast<const uint4*>(p + 16);
          b[ni] = un.v;
        }
#pragma unroll
        for (int mi = 0; mi < 4; ++mi) {
          union { v16bf v; uint4 u[2]; } un;
          const unsigned short* p = &Atile[m_w * 64 + mi * 16 + mrow][koff];
          un.u[0] = *reinterpret_cast<const uint4*>(p);
          un.u[1] = *reinterpret_cast<const uint4*>(p + 16);
          v16bf a = un.v;
#pragma unroll
          for (int ni = 0; ni < 4; ++ni)
            acc[mi][ni] = __builtin_amdgcn_wmma_f32_16x16x32_bf16(
                false, a, false, b[ni], (short)0, acc[mi][ni], false, false);
        }
      }
    }

    // ---- selection in 4 column passes of 64 ------------------------------
#pragma unroll 1
    for (int p = 0; p < NSTEP / 64; ++p) {
      __syncthreads();   // previous pass reads (or K-loop frag reads) done
      if (n_w == p) {
#pragma unroll
        for (int mi = 0; mi < 4; ++mi)
#pragma unroll
          for (int ni = 0; ni < 4; ++ni)
#pragma unroll
            for (int v = 0; v < 8; ++v) {
              const int m = m_w * 64 + mi * 16 + (lane >> 4) * 8 + v;
              const int n = ni * 16 + (lane & 15);
              Stile[m][n] = acc[mi][ni][v];
            }
      }
      __syncthreads();
      for (int c = 0; c < 32; ++c) {
        const int cc   = selHalf * 32 + c;
        const int gcol = nbase + p * 64 + cc;
        if (gcol < NM) top4_insert(bv, bi, Stile[selRow][cc], gcol);
      }
    }
  }

  // ---- merge the two column-halves of each row, emit chunk partials ------
  __syncthreads();
  float* sval  = &Stile[0][0];
  int*   sidx2 = reinterpret_cast<int*>(sval + QT * 8);
  if (selHalf) {
    const int base = selRow * TOPK;
#pragma unroll
    for (int j = 0; j < TOPK; ++j) { sval[base + j] = bv[j]; sidx2[base + j] = bi[j]; }
  }
  __syncthreads();
  if (!selHalf) {
    const int base = selRow * TOPK;
#pragma unroll
    for (int j = 0; j < TOPK; ++j) top4_insert(bv, bi, sval[base + j], sidx2[base + j]);
    const int prow = qrow0 + selRow;
    const size_t off = ((size_t)prow * NCHUNK + blockIdx.y) * TOPK;
#pragma unroll
    for (int j = 0; j < TOPK; ++j) { pval[off + j] = bv[j]; pidx[off + j] = bi[j]; }
  }
}

// -------------------------------------------- final merge + kNN average ---
__global__ __launch_bounds__(256) void gather_kernel(
    const float* __restrict__ synth, const float* __restrict__ pval,
    const int* __restrict__ pidx, float* __restrict__ out) {
  __shared__ int sidx[TOPK];
  const int row = blockIdx.x;
  if (threadIdx.x == 0) {
    float bv[TOPK] = {NEG_INF, NEG_INF, NEG_INF, NEG_INF};
    int   bi[TOPK] = {0, 0, 0, 0};
    const float* pv = pval + (size_t)row * NCHUNK * TOPK;
    const int*   pi = pidx + (size_t)row * NCHUNK * TOPK;
    for (int e = 0; e < NCHUNK * TOPK; ++e) top4_insert(bv, bi, pv[e], pi[e]);
#pragma unroll
    for (int j = 0; j < TOPK; ++j) sidx[j] = bi[j];
  }
  __syncthreads();
  const float* s0 = synth + (size_t)sidx[0] * DIM;
  const float* s1 = synth + (size_t)sidx[1] * DIM;
  const float* s2 = synth + (size_t)sidx[2] * DIM;
  const float* s3 = synth + (size_t)sidx[3] * DIM;
  for (int d = threadIdx.x; d < DIM; d += blockDim.x)
    out[(size_t)row * DIM + d] = 0.25f * (s0[d] + s1[d] + s2[d] + s3[d]);
}

// --------------------------------------------------------------- launch ---
extern "C" void kernel_launch(void* const* d_in, const int* in_sizes, int n_in,
                              void* d_out, int out_size, void* d_ws, size_t ws_size,
                              hipStream_t stream) {
  const float* Q = (const float*)d_in[0];   // [2048, 1024]
  const float* M = (const float*)d_in[1];   // [100000, 1024]
  const float* S = (const float*)d_in[2];   // [100000, 1024]
  // d_in[3] = topk (fixed at 4 in the reference)
  float* out = (float*)d_out;

  char* ws = (char*)d_ws;
  const size_t OFF_Q    = 0;                                   // 2048*1024*2   = 4 MB
  const size_t OFF_M    = (size_t)NQ * DIM * 2;                // 100000*1024*2 = 204.8 MB
  const size_t OFF_PVAL = OFF_M + (size_t)NM * DIM * 2;
  const size_t SZ_P     = (size_t)NQ * NCHUNK * TOPK * 4;      // 1,605,632 B
  unsigned short* Qb = (unsigned short*)(ws + OFF_Q);
  unsigned short* Mb = (unsigned short*)(ws + OFF_M);
  float* pval = (float*)(ws + OFF_PVAL);
  int*   pidx = (int*)(ws + OFF_PVAL + SZ_P);

  hipLaunchKernelGGL(conv_q_kernel, dim3(NQ / 8), dim3(256), 0, stream, Q, Qb);
  hipLaunchKernelGGL(conv_m_kernel, dim3((NM + 7) / 8), dim3(256), 0, stream, M, Mb);
  hipLaunchKernelGGL(simtopk_kernel, dim3(NQ / QT, NCHUNK), dim3(256), 0, stream,
                     Qb, Mb, pval, pidx);
  hipLaunchKernelGGL(gather_kernel, dim3(NQ), dim3(256), 0, stream,
                     S, pval, pidx, out);
}
